// SelectiveSSMBlock_26534307954721
// MI455X (gfx1250) — matrix operand order repaired and checked by reading, not compiled, not run on hardware
//
#include <hip/hip_runtime.h>

// ---------------------------------------------------------------------------
// SelectiveSSMBlock for MI455X (gfx1250, wave32)
//   B=2, L=2048, D_MODEL=1024, D_STATE=16, DT_RANK=64
//   - GEMMs: v_wmma_f32_16x16x32_f16 (f16 in, f32 accumulate)
//   - GEMM1: one wave per 16x96 row panel, f32 x converted in-register
//   - GEMM2: weight panel staged to LDS via TDM (tensor_load_to_lds)
//   - Scan: 3-phase chunked parallel scan (16 chunks x 128 steps)
// ---------------------------------------------------------------------------

#define BB      2
#define LL      2048
#define DM      1024
#define NS      16          // D_STATE
#define DR      64          // DT_RANK
#define NPROJ   96          // DT_RANK + 2*D_STATE
#define ROWS    (BB * LL)   // 4096
#define NC      16          // scan chunks
#define CL      (LL / NC)   // 128 steps per chunk

typedef _Float16 v16h __attribute__((ext_vector_type(16)));
typedef _Float16 v8h  __attribute__((ext_vector_type(8)));
typedef float    v8f  __attribute__((ext_vector_type(8)));
typedef unsigned v4u  __attribute__((ext_vector_type(4)));
typedef int      v8i  __attribute__((ext_vector_type(8)));
typedef int      v4i  __attribute__((ext_vector_type(4)));

// ---------------- workspace layout (bytes) ----------------
#define OFF_W1H    0               // NPROJ*DM f16 (scaled weight)   192 KiB
#define OFF_W2H    196608          // DM*DR f16                      128 KiB
#define OFF_XZ     327680          // ROWS*NPROJ f32                 1.5 MiB
#define OFF_DXH    1900544         // ROWS*DR f16                    512 KiB
#define OFF_DELTA  2424832         // ROWS*DM f32                    16 MiB
#define OFF_APROD  19202048        // BB*NC*DM*NS f32                2 MiB
#define OFF_HLOC   21299200        // BB*NC*DM*NS f32                2 MiB
#define OFF_HIN    23396352        // BB*NC*DM*NS f32                2 MiB

// ---------------------------------------------------------------------------
// Prep kernels (tiny)
// ---------------------------------------------------------------------------
// w1h[n][k] = f16( x_proj_w[n][k] * sigmoid(trend[k]) )  -- fold the activation
// scale into the 96x1024 weight instead of rescaling the 16 MB activation.
__global__ void prep_w1(const float* __restrict__ w, const float* __restrict__ trend,
                        _Float16* __restrict__ w1h) {
    int i = blockIdx.x * blockDim.x + threadIdx.x;   // NPROJ*DM threads
    int k = i & (DM - 1);
    float s = 1.0f / (1.0f + __expf(-trend[k]));
    w1h[i] = (_Float16)(w[i] * s);
}

__global__ void cvt_w2_f16(const float* __restrict__ w, _Float16* __restrict__ w2h) {
    int i = blockIdx.x * blockDim.x + threadIdx.x;   // DM*DR threads
    w2h[i] = (_Float16)w[i];
}

// ---------------------------------------------------------------------------
// Tensor Data Mover: stage a 2D f16 tile (tile_d1 rows x tile_d0 cols) into LDS.
// D# per CDNA5 ISA (2D: groups 2/3 zero / NULL-like).
// ---------------------------------------------------------------------------
__device__ __forceinline__ void tdm_load_2d_f16(unsigned lds_off, const void* gptr,
                                                unsigned tensor_d0, unsigned tensor_d1,
                                                unsigned tile_d0, unsigned tile_d1,
                                                unsigned row_stride) {
    unsigned long long ga = (unsigned long long)gptr;
    v4u g0;
    g0[0] = 1u;                                   // count=1, user descriptor
    g0[1] = lds_off;                              // lds_addr (bytes)
    g0[2] = (unsigned)(ga & 0xFFFFFFFFu);         // global_addr[31:0]
    g0[3] = (unsigned)((ga >> 32) & 0x1FFFFFFu)   // global_addr[56:32]
          | (2u << 30);                           // type = 2 ("image")
    v8i g1;
    g1[0] = 0x00010000;                           // wg_mask=0, data_size=1 (2B)
    g1[1] = (int)((tensor_d0 & 0xFFFFu) << 16);   // tensor_dim0[15:0] @ bits63:48
    g1[2] = (int)((tensor_d0 >> 16) & 0xFFFFu)    // tensor_dim0[31:16]
          | (int)((tensor_d1 & 0xFFFFu) << 16);   // tensor_dim1[15:0]
    g1[3] = (int)((tensor_d1 >> 16) & 0xFFFFu)    // tensor_dim1[31:16]
          | (int)((tile_d0 & 0xFFFFu) << 16);     // tile_dim0
    g1[4] = (int)(tile_d1 & 0xFFFFu);             // tile_dim1 (tile_dim2 = 0)
    g1[5] = (int)row_stride;                      // tensor_dim0_stride[31:0]
    g1[6] = 0;
    g1[7] = 0;
    v4i z4 = {0, 0, 0, 0};
#if defined(__clang_major__) && (__clang_major__ >= 23)
    v8i z8 = {0, 0, 0, 0, 0, 0, 0, 0};
    __builtin_amdgcn_tensor_load_to_lds(g0, g1, z4, z4, z8, 0);
#else
    __builtin_amdgcn_tensor_load_to_lds(g0, g1, z4, z4, 0);
#endif
}

// ---------------------------------------------------------------------------
// GEMM1: xz[ROWS x 96] = scaled_x @ x_proj_w^T.
// One wave per 16-row panel computes all six 16x16 column tiles: the f32 x row
// is read ONCE and converted to f16 in-register. Columns 0..63 also written as
// f16 (dxh) to feed GEMM2's A operand.
// grid = ROWS/64 blocks, block = 128 (4 waves, one panel each).
// ---------------------------------------------------------------------------
__global__ void gemm1_wmma(const float* __restrict__ x,
                           const _Float16* __restrict__ w1h,
                           float* __restrict__ xz,
                           _Float16* __restrict__ dxh) {
    const int wave = threadIdx.x >> 5;
    const int lane = threadIdx.x & 31;
    const int h    = lane >> 4;
    const int mn   = lane & 15;
    const int rb   = (blockIdx.x * 4 + wave) * 16;

    const float* arow = x + (size_t)(rb + mn) * DM;

    v8f acc[6] = {};
    for (int k0 = 0; k0 < DM; k0 += 32) {
        // A frag: lane half h holds K = k0+16h..+15; cvt f32->f16.
        v16h a;
#pragma unroll
        for (int i = 0; i < 8; ++i) {
            a[i]     = (_Float16)arow[k0 + 16 * h + i];
            a[8 + i] = (_Float16)arow[k0 + 16 * h + 8 + i];
        }
#pragma unroll
        for (int t = 0; t < 6; ++t) {
            // B frag: lanes 0-15 hold K=k0..k0+15, lanes 16-31 hold K=k0+16..+31
            v16h b = *(const v16h*)(w1h + (size_t)(t * 16 + mn) * DM + k0 + 16 * h);
            acc[t] = __builtin_amdgcn_wmma_f32_16x16x32_f16(
                false, a, false, b, (short)0, acc[t], false, false);
        }
    }

    // D layout: VGPR r -> row rb + r + 8h, col = 16t + (lane&15)
    const int row0 = rb + 8 * h;
#pragma unroll
    for (int t = 0; t < 6; ++t) {
        const int col = t * 16 + mn;
#pragma unroll
        for (int r = 0; r < 8; ++r)
            xz[(size_t)(row0 + r) * NPROJ + col] = acc[t][r];
    }
#pragma unroll
    for (int t = 0; t < 4; ++t) {              // cols 0..63 -> f16 copy for GEMM2
        const int col = t * 16 + mn;
#pragma unroll
        for (int r = 0; r < 8; ++r)
            dxh[(size_t)(row0 + r) * DR + col] = (_Float16)acc[t][r];
    }
}

// ---------------------------------------------------------------------------
// GEMM2 + bias + softplus:
//   delta[ROWS x 1024] = softplus( dxh[ROWS x 64] @ dt_proj_w^T + b )
// grid = (4 column-quarters, ROWS/16 row panels), block = 128 (4 waves).
// The 256x64 f16 weight panel (32 KiB) is staged into LDS by the TDM; each
// wave then computes four 16x16 tiles, A-frags held in registers across tiles.
// ---------------------------------------------------------------------------
__global__ void gemm2_wmma(const _Float16* __restrict__ dxh,
                           const _Float16* __restrict__ w2h,
                           const float* __restrict__ bias,
                           float* __restrict__ delta) {
    __shared__ _Float16 w2s[256 * DR];            // 32 KiB

    const int cq   = blockIdx.x;                  // column quarter: cols [cq*256, +256)
    const int rb   = blockIdx.y * 16;
    const int wave = threadIdx.x >> 5;
    const int lane = threadIdx.x & 31;
    const int h    = lane >> 4;
    const int mn   = lane & 15;

    if (wave == 0) {
        tdm_load_2d_f16((unsigned)(uintptr_t)&w2s[0],
                        w2h + (size_t)cq * 256 * DR,
                        /*tensor_d0=*/DR, /*tensor_d1=*/DM,
                        /*tile_d0=*/DR, /*tile_d1=*/256,
                        /*row_stride=*/DR);
        __builtin_amdgcn_s_wait_tensorcnt(0);
    }
    __syncthreads();

    // A frags for K chunks 0 and 32, reused across all four column tiles.
    const _Float16* arow = dxh + (size_t)(rb + mn) * DR;
    v16h a0, a1;
    {
        v8h lo0 = *(const v8h*)(arow + 16 * h);
        v8h hi0 = *(const v8h*)(arow + 8 + 16 * h);
        v8h lo1 = *(const v8h*)(arow + 32 + 16 * h);
        v8h hi1 = *(const v8h*)(arow + 40 + 16 * h);
#pragma unroll
        for (int i = 0; i < 8; ++i) {
            a0[i] = lo0[i]; a0[8 + i] = hi0[i];
            a1[i] = lo1[i]; a1[8 + i] = hi1[i];
        }
    }

    const int row0 = rb + 8 * h;
#pragma unroll
    for (int t = 0; t < 4; ++t) {
        const int lrow = (wave * 4 + t) * 16 + mn;       // row within staged panel
        v16h b0 = *(const v16h*)(w2s + (size_t)lrow * DR + 16 * h);
        v16h b1 = *(const v16h*)(w2s + (size_t)lrow * DR + 32 + 16 * h);
        v8f acc = {};
        acc = __builtin_amdgcn_wmma_f32_16x16x32_f16(
            false, a0, false, b0, (short)0, acc, false, false);
        acc = __builtin_amdgcn_wmma_f32_16x16x32_f16(
            false, a1, false, b1, (short)0, acc, false, false);

        const int col = cq * 256 + lrow - mn + mn;       // = cq*256 + (wave*4+t)*16 + mn
        const float bv = bias[col];
#pragma unroll
        for (int r = 0; r < 8; ++r) {
            float v  = acc[r] + bv;
            float sp = (v > 20.0f) ? v : log1pf(__expf(v));   // stable softplus
            delta[(size_t)(row0 + r) * DM + col] = sp;
        }
    }
}

// ---------------------------------------------------------------------------
// Chunked parallel scan. h_t = a_t h_{t-1} + b_t is linear, so:
//  phase1: per (b,d,chunk) compute local end-state (zero init) + decay product
//  phase2: per (b,d,n) 16-step chunk prefix -> incoming state per chunk
//  phase3: replay each chunk with correct incoming state, emit y
// Critical path: 128 + 16 + 128 dependent steps instead of 2048.
// ---------------------------------------------------------------------------
__global__ void scan_phase1(const float* __restrict__ x,
                            const float* __restrict__ A_log,
                            const float* __restrict__ trend,
                            const float* __restrict__ xz,
                            const float* __restrict__ delta,
                            float* __restrict__ aprod,
                            float* __restrict__ hloc) {
    const int t  = blockIdx.x * 256 + threadIdx.x;   // 0 .. BB*NC*DM-1
    const int d  = t & (DM - 1);
    const int bc = t >> 10;                          // b*NC + c (uniform per block)
    const int b  = bc >> 4, c = bc & (NC - 1);

    const float s = 1.0f / (1.0f + __expf(-trend[d]));
    float A[NS], hs[NS], ap[NS];
    const float4* Arow = (const float4*)(A_log + (size_t)d * NS);
#pragma unroll
    for (int q = 0; q < 4; ++q) {
        float4 aq = Arow[q];
        A[4 * q + 0] = -__expf(aq.x); A[4 * q + 1] = -__expf(aq.y);
        A[4 * q + 2] = -__expf(aq.z); A[4 * q + 3] = -__expf(aq.w);
    }
#pragma unroll
    for (int n = 0; n < NS; ++n) { hs[n] = 0.0f; ap[n] = 1.0f; }

    const size_t base = (size_t)b * LL + (size_t)c * CL;
    for (int l = 0; l < CL; ++l) {
        const size_t idx = base + l;
        const size_t pidx = base + (l + 8 < CL ? l + 8 : CL - 1);
        __builtin_prefetch(&delta[pidx * DM + d], 0, 0);
        __builtin_prefetch(&x[pidx * DM + d], 0, 0);

        const float dlt = delta[idx * DM + d];
        const float u   = x[idx * DM + d] * s;
        const float du  = dlt * u;
        const float4* B4 = (const float4*)(xz + idx * NPROJ + DR);  // uniform addr
        float Bv[NS];
#pragma unroll
        for (int q = 0; q < 4; ++q) {
            float4 bq = B4[q];
            Bv[4 * q] = bq.x; Bv[4 * q + 1] = bq.y; Bv[4 * q + 2] = bq.z; Bv[4 * q + 3] = bq.w;
        }
#pragma unroll
        for (int n = 0; n < NS; ++n) {
            float dA = __expf(dlt * A[n]);
            ap[n] *= dA;
            hs[n] = fmaf(dA, hs[n], du * Bv[n]);
        }
    }
    float4* apo = (float4*)(aprod + ((size_t)bc * DM + d) * NS);
    float4* hlo = (float4*)(hloc  + ((size_t)bc * DM + d) * NS);
#pragma unroll
    for (int q = 0; q < 4; ++q) {
        apo[q] = make_float4(ap[4 * q], ap[4 * q + 1], ap[4 * q + 2], ap[4 * q + 3]);
        hlo[q] = make_float4(hs[4 * q], hs[4 * q + 1], hs[4 * q + 2], hs[4 * q + 3]);
    }
}

__global__ void scan_phase2(const float* __restrict__ aprod,
                            const float* __restrict__ hloc,
                            float* __restrict__ hin) {
    const int q = blockIdx.x * 256 + threadIdx.x;    // 0 .. BB*DM*NS-1
    const int n = q & (NS - 1);
    const int d = (q >> 4) & (DM - 1);
    const int b = q >> 14;
    float hcur = 0.0f;
#pragma unroll
    for (int c = 0; c < NC; ++c) {
        const size_t off = ((size_t)(b * NC + c) * DM + d) * NS + n;
        hin[off] = hcur;                              // state entering chunk c
        hcur = fmaf(aprod[off], hcur, hloc[off]);
    }
}

__global__ void scan_phase3(const float* __restrict__ x,
                            const float* __restrict__ A_log,
                            const float* __restrict__ trend,
                            const float* __restrict__ D_param,
                            const float* __restrict__ xz,
                            const float* __restrict__ delta,
                            const float* __restrict__ hin,
                            float* __restrict__ out) {
    const int t  = blockIdx.x * 256 + threadIdx.x;
    const int d  = t & (DM - 1);
    const int bc = t >> 10;
    const int b  = bc >> 4, c = bc & (NC - 1);

    const float s  = 1.0f / (1.0f + __expf(-trend[d]));
    const float Dp = D_param[d];
    float A[NS], hs[NS];
    const float4* Arow = (const float4*)(A_log + (size_t)d * NS);
#pragma unroll
    for (int q = 0; q < 4; ++q) {
        float4 aq = Arow[q];
        A[4 * q + 0] = -__expf(aq.x); A[4 * q + 1] = -__expf(aq.y);
        A[4 * q + 2] = -__expf(aq.z); A[4 * q + 3] = -__expf(aq.w);
    }
    const float4* h4 = (const float4*)(hin + ((size_t)bc * DM + d) * NS);
#pragma unroll
    for (int q = 0; q < 4; ++q) {
        float4 hq = h4[q];
        hs[4 * q] = hq.x; hs[4 * q + 1] = hq.y; hs[4 * q + 2] = hq.z; hs[4 * q + 3] = hq.w;
    }

    const size_t base = (size_t)b * LL + (size_t)c * CL;
    for (int l = 0; l < CL; ++l) {
        const size_t idx = base + l;
        const size_t pidx = base + (l + 8 < CL ? l + 8 : CL - 1);
        __builtin_prefetch(&delta[pidx * DM + d], 0, 0);
        __builtin_prefetch(&x[pidx * DM + d], 0, 0);

        const float dlt = delta[idx * DM + d];
        const float u   = x[idx * DM + d] * s;
        const float du  = dlt * u;
        const float4* B4 = (const float4*)(xz + idx * NPROJ + DR);
        const float4* C4 = (const float4*)(xz + idx * NPROJ + DR + NS);
        float Bv[NS], Cv[NS];
#pragma unroll
        for (int q = 0; q < 4; ++q) {
            float4 bq = B4[q], cq = C4[q];
            Bv[4 * q] = bq.x; Bv[4 * q + 1] = bq.y; Bv[4 * q + 2] = bq.z; Bv[4 * q + 3] = bq.w;
            Cv[4 * q] = cq.x; Cv[4 * q + 1] = cq.y; Cv[4 * q + 2] = cq.z; Cv[4 * q + 3] = cq.w;
        }
        float y = 0.0f;
#pragma unroll
        for (int n = 0; n < NS; ++n) {
            float dA = __expf(dlt * A[n]);
            hs[n] = fmaf(dA, hs[n], du * Bv[n]);
            y = fmaf(hs[n], Cv[n], y);
        }
        out[idx * DM + d] = fmaf(u, Dp, y);
    }
}

// ---------------------------------------------------------------------------
extern "C" void kernel_launch(void* const* d_in, const int* in_sizes, int n_in,
                              void* d_out, int out_size, void* d_ws, size_t ws_size,
                              hipStream_t stream) {
    const float* x         = (const float*)d_in[0];   // (2,2048,1024)
    const float* A_log     = (const float*)d_in[1];   // (1024,16)
    const float* x_proj_w  = (const float*)d_in[2];   // (96,1024)
    const float* dt_proj_w = (const float*)d_in[3];   // (1024,64)
    const float* dt_proj_b = (const float*)d_in[4];   // (1024,)
    const float* D_param   = (const float*)d_in[5];   // (1024,)
    const float* trend     = (const float*)d_in[6];   // (1024,)
    float* out = (float*)d_out;

    char* ws = (char*)d_ws;
    _Float16* w1h   = (_Float16*)(ws + OFF_W1H);
    _Float16* w2h   = (_Float16*)(ws + OFF_W2H);
    float*    xz    = (float*)   (ws + OFF_XZ);
    _Float16* dxh   = (_Float16*)(ws + OFF_DXH);
    float*    delta = (float*)   (ws + OFF_DELTA);
    float*    aprod = (float*)   (ws + OFF_APROD);
    float*    hloc  = (float*)   (ws + OFF_HLOC);
    float*    hin   = (float*)   (ws + OFF_HIN);

    prep_w1   <<<(NPROJ * DM) / 256, 256, 0, stream>>>(x_proj_w, trend, w1h);
    cvt_w2_f16<<<(DM * DR) / 256,    256, 0, stream>>>(dt_proj_w, w2h);

    // xz = scaled_x @ x_proj_w^T   (WMMA row panels, x read once)
    gemm1_wmma<<<ROWS / 64, 128, 0, stream>>>(x, w1h, xz, dxh);

    // delta = softplus(xz[:, :64] @ dt_proj_w^T + b)   (WMMA, TDM-staged B)
    gemm2_wmma<<<dim3(4, ROWS / 16), 128, 0, stream>>>(dxh, w2h, dt_proj_b, delta);

    // chunked parallel selective scan
    scan_phase1<<<(BB * NC * DM) / 256, 256, 0, stream>>>(x, A_log, trend, xz, delta,
                                                          aprod, hloc);
    scan_phase2<<<(BB * DM * NS) / 256, 256, 0, stream>>>(aprod, hloc, hin);
    scan_phase3<<<(BB * NC * DM) / 256, 256, 0, stream>>>(x, A_log, trend, D_param,
                                                          xz, delta, hin, out);
}
